// CNNseq_15564961481149
// MI455X (gfx1250) — matrix operand order repaired and checked by reading, not compiled, not run on hardware
//
#include <hip/hip_runtime.h>
#include <hip/hip_bf16.h>
#include <stdint.h>

// ---------------------------------------------------------------------------
// TextCNN: embed-gather -> conv1d(K=3,4,5, F=256, D=512) -> bias -> max-over-t
//          -> relu -> concat.  bf16 WMMA (v_wmma_f32_16x16x32_bf16), f32 acc.
// v2: fuse both M-tiles per wave into the inner loop (2 WMMAs per B-tile) and
//     double-buffer B from LDS to break the ds_load -> wait -> wmma chain.
// ---------------------------------------------------------------------------

typedef __attribute__((ext_vector_type(16))) __bf16        v16bf;
typedef __attribute__((ext_vector_type(8)))  float         v8f;
typedef __attribute__((ext_vector_type(4)))  unsigned int  u32x4;

union Pack32 { u32x4 q[2]; v16bf v; };   // 32 bytes = 16 bf16 = one WMMA operand

#define DCH        512      // embedding dim
#define LSEQ       128      // sequence length
#define NF         256      // filters per kernel size
#define VOCAB      30000
#define XROWS      132      // 128 + 4 zero-pad rows (t+k overrun)
#define XSTRIDE_B  1040     // bytes per LDS row: (512 + 8 pad) * 2
#define BS_TOTAL   1024     // 16 * 64

// ---- one-time f32 -> bf16 conversion of the embedding table ----
__global__ void f32_to_bf16_kernel(const float* __restrict__ in,
                                   __hip_bfloat16* __restrict__ out, int n) {
    int i = blockIdx.x * blockDim.x + threadIdx.x;
    if (i < n) out[i] = __float2bfloat16(in[i]);
}

// ---- repack conv weights [F, D, K] f32 -> [sliceBase+k][F][D] bf16 ----
__global__ void pack_w_kernel(const float* __restrict__ w,
                              __hip_bfloat16* __restrict__ out,
                              int K, int sliceBase) {
    int i = blockIdx.x * blockDim.x + threadIdx.x;
    int n = NF * DCH * K;
    if (i >= n) return;
    int k = i % K;
    int d = (i / K) % DCH;
    int f = i / (K * DCH);
    out[(size_t)(sliceBase + k) * NF * DCH + (size_t)f * DCH + d] =
        __float2bfloat16(w[i]);
}

// ---- max-over-time + bias + relu for one 16-filter tile ----
// C layout: lane holds column N = lm; VGPR r -> row M = r + 8*lh.
__device__ __forceinline__
void pool_store(const v8f (&acc)[8], int Lout, int lm, int lh,
                int fbase, const float* __restrict__ bias,
                float* __restrict__ outp) {
    #pragma unroll
    for (int r = 0; r < 8; ++r) {
        float v = -1e30f;
        #pragma unroll
        for (int nt = 0; nt < 8; ++nt) {
            const int t = nt * 16 + lm;
            const float cand = (t < Lout) ? acc[nt][r] : -1e30f;
            v = fmaxf(v, cand);
        }
        v = fmaxf(v, __shfl_xor(v, 1, 16));
        v = fmaxf(v, __shfl_xor(v, 2, 16));
        v = fmaxf(v, __shfl_xor(v, 4, 16));
        v = fmaxf(v, __shfl_xor(v, 8, 16));
        if (lm == 0) {
            const int f = fbase + lh * 8 + r;
            outp[f] = fmaxf(v + bias[f], 0.0f);
        }
    }
}

// ---- main kernel: one workgroup per bs sample, 8 waves, 2 M-tiles/wave ----
__global__ __launch_bounds__(256)
void cnnpool_kernel(const int* __restrict__ text,
                    const __hip_bfloat16* __restrict__ embed,   // bf16 [V][D]
                    const __hip_bfloat16* __restrict__ wsl,     // bf16 [12][F][D]
                    const float* __restrict__ b3,
                    const float* __restrict__ b4,
                    const float* __restrict__ b5,
                    float* __restrict__ out) {
    extern __shared__ char smem[];           // XROWS rows * XSTRIDE_B bytes
    const int tid = threadIdx.x;
    const int bs  = blockIdx.x;

    // Phase 1: gather 128 embedding rows (bf16) into LDS. 2 threads/row.
    {
        const int r    = tid >> 1;           // 0..127
        const int half = tid & 1;            // each copies 256 bf16 = 512 B
        const int tok  = text[bs * LSEQ + r];
        const u32x4* src = (const u32x4*)(embed + (size_t)tok * DCH + half * 256);
        char* dstRow = smem + r * XSTRIDE_B + half * 512;
        #pragma unroll
        for (int j = 0; j < 32; ++j)
            *(u32x4*)(dstRow + j * 16) = src[j];
    }
    // zero the 4 pad rows (rows 128..131): 4*1040 B = 260 x 16 B
    for (int i = tid; i < 260; i += 256) {
        u32x4 z = {0u, 0u, 0u, 0u};
        *(u32x4*)(smem + 128 * XSTRIDE_B + i * 16) = z;
    }
    __syncthreads();

    const int lane = tid & 31;
    const int wave = tid >> 5;
    const int lh   = lane >> 4;              // lane half (K-split in A/B layout)
    const int lm   = lane & 15;

    for (int c = 0; c < 3; ++c) {            // kernel sizes 3, 4, 5
        const int K    = 3 + c;
        const int sb   = (c == 0) ? 0 : (c == 1) ? 3 : 7;  // weight slice base
        const int Lout = LSEQ - K + 1;
        const float* bias = (c == 0) ? b3 : (c == 1) ? b4 : b5;

        // wave owns filter rows [wave*32, wave*32+32): two 16-row M-tiles.
        v8f acc0[8], acc1[8];
        const v8f vzero = {};
        #pragma unroll
        for (int nt = 0; nt < 8; ++nt) { acc0[nt] = vzero; acc1[nt] = vzero; }

        for (int k = 0; k < K; ++k) {
            const __hip_bfloat16* w0 =
                wsl + (size_t)(sb + k) * NF * DCH
                    + (size_t)(wave * 32 + lm) * DCH;
            const __hip_bfloat16* w1 = w0 + 16 * DCH;

            for (int ch = 0; ch < 16; ++ch) {   // d chunks of 32
                const int dbase = ch * 32;
                // Two A tiles (this wave's pair of M-tiles), reused over 8 nt.
                Pack32 a0, a1;
                a0.q[0] = *(const u32x4*)(w0 + dbase + lh * 8);
                a0.q[1] = *(const u32x4*)(w0 + dbase + 16 + lh * 8);
                a1.q[0] = *(const u32x4*)(w1 + dbase + lh * 8);
                a1.q[1] = *(const u32x4*)(w1 + dbase + 16 + lh * 8);

                // B from LDS, double-buffered: each B feeds 2 WMMAs.
                const char* bbase =
                    smem + (lm + k) * XSTRIDE_B + dbase * 2 + lh * 16;
                Pack32 bcur;
                bcur.q[0] = *(const u32x4*)(bbase);
                bcur.q[1] = *(const u32x4*)(bbase + 32);
                #pragma unroll
                for (int nt = 0; nt < 8; ++nt) {
                    Pack32 bnxt = bcur;
                    if (nt < 7) {
                        const char* p = bbase + (nt + 1) * 16 * XSTRIDE_B;
                        bnxt.q[0] = *(const u32x4*)(p);
                        bnxt.q[1] = *(const u32x4*)(p + 32);
                    }
                    acc0[nt] = __builtin_amdgcn_wmma_f32_16x16x32_bf16(
                        false, a0.v, false, bcur.v,
                        (short)0, acc0[nt], false, false);
                    acc1[nt] = __builtin_amdgcn_wmma_f32_16x16x32_bf16(
                        false, a1.v, false, bcur.v,
                        (short)0, acc1[nt], false, false);
                    bcur = bnxt;
                }
            }
        }

        float* outp = out + (size_t)bs * 768 + c * 256;
        pool_store(acc0, Lout, lm, lh, wave * 32,      bias, outp);
        pool_store(acc1, Lout, lm, lh, wave * 32 + 16, bias, outp);
    }
}

extern "C" void kernel_launch(void* const* d_in, const int* in_sizes, int n_in,
                              void* d_out, int out_size, void* d_ws, size_t ws_size,
                              hipStream_t stream) {
    (void)in_sizes; (void)n_in; (void)out_size; (void)ws_size;

    const int*   text  = (const int*)d_in[0];
    const float* embed = (const float*)d_in[1];
    const float* w3    = (const float*)d_in[2];
    const float* b3    = (const float*)d_in[3];
    const float* w4    = (const float*)d_in[4];
    const float* b4    = (const float*)d_in[5];
    const float* w5    = (const float*)d_in[6];
    const float* b5    = (const float*)d_in[7];
    float*       out   = (float*)d_out;

    // workspace layout: [embed bf16: V*D*2 = 30,720,000 B][weights bf16: 12*F*D*2]
    __hip_bfloat16* embed_bf = (__hip_bfloat16*)d_ws;
    __hip_bfloat16* wsl =
        (__hip_bfloat16*)((char*)d_ws + (size_t)VOCAB * DCH * sizeof(__hip_bfloat16));

    const int nE = VOCAB * DCH;
    f32_to_bf16_kernel<<<(nE + 255) / 256, 256, 0, stream>>>(embed, embed_bf, nE);
    pack_w_kernel<<<(NF * DCH * 3 + 255) / 256, 256, 0, stream>>>(w3, wsl, 3, 0);
    pack_w_kernel<<<(NF * DCH * 4 + 255) / 256, 256, 0, stream>>>(w4, wsl, 4, 3);
    pack_w_kernel<<<(NF * DCH * 5 + 255) / 256, 256, 0, stream>>>(w5, wsl, 5, 7);

    const size_t lds_bytes = (size_t)XROWS * XSTRIDE_B;   // 137,280 B
    hipFuncSetAttribute((const void*)cnnpool_kernel,
                        hipFuncAttributeMaxDynamicSharedMemorySize,
                        (int)lds_bytes);
    cnnpool_kernel<<<BS_TOTAL, 256, lds_bytes, stream>>>(
        text, embed_bf, wsl, b3, b4, b5, out);
}
